// MixtureOfExperts_60799557042406
// MI455X (gfx1250) — matrix operand order repaired and checked by reading, not compiled, not run on hardware
//
#include <hip/hip_runtime.h>
#include <hip/hip_bf16.h>
#include <cmath>

// Problem constants: B=2, S=1024, D=1024, H=4096, O=1024, E=8, TOP_K=2
#define D_    1024
#define H_    4096
#define O_    1024
#define E_    8
#define NTOK  2048
#define TOPK  2

// Expert-GEMM tiling
#define TM    16            // tokens per tile (WMMA M)
#define TN    256           // output cols per iter (8 waves x 2 tiles x 16)
#define TK    32            // WMMA K per step (bf16 k=32)
#define TILES_PER_E 128

// LDS strides (bf16 elements); row strides multiples of 8 elems (16B) for b128 ds ops.
#define XS_LD 1032
#define HS_LD 4104
#define BS_LD 40
#define BS_BUF (TN * BS_LD)
#define LDS_BYTES_STAGE (TM*XS_LD*2 + TM*HS_LD*2 + 2*BS_BUF*2 + TM*4 + TM*4) // 205440
#define LDS_BYTES_FAST  (TM*XS_LD*2 + TM*HS_LD*2 + TM*4 + TM*4)              // 164480

// Workspace layout
#define WS_LIST_OFF  64
#define WS_WLIST_OFF (WS_LIST_OFF + E_*NTOK*4)
#define WS_W1T_OFF   262144ull                              // 256KB aligned
#define WS_W2T_OFF   (WS_W1T_OFF + (size_t)E_*D_*H_*2)
#define WS_NEED      (WS_W2T_OFF + (size_t)E_*H_*O_*2)      // ~128MB + lists

typedef float          v8f   __attribute__((ext_vector_type(8)));
typedef __bf16         v16bf __attribute__((ext_vector_type(16)));
typedef __bf16         v2bf  __attribute__((ext_vector_type(2)));
typedef unsigned short v8u16 __attribute__((ext_vector_type(8)));

static __device__ __forceinline__ unsigned short f2bf(float f) {
  unsigned int u = __builtin_bit_cast(unsigned int, f);
  return (unsigned short)((u + 0x8000u) >> 16);
}

static __device__ __forceinline__ unsigned int pk_bf16(float a, float b) {
#if __has_builtin(__builtin_amdgcn_cvt_pk_bf16_f32)
  v2bf t = __builtin_amdgcn_cvt_pk_bf16_f32(a, b);
  return __builtin_bit_cast(unsigned int, t);
#else
  unsigned int ua = __builtin_bit_cast(unsigned int, a) + 0x8000u;
  unsigned int ub = __builtin_bit_cast(unsigned int, b) + 0x8000u;
  return __builtin_amdgcn_perm(ub, ua, 0x07060302u);   // {ub[31:16], ua[31:16]}
#endif
}

struct frag_pair { v8u16 lo, hi; };
static __device__ __forceinline__ v16bf make_frag(v8u16 lo, v8u16 hi) {
  frag_pair p{lo, hi};
  return __builtin_bit_cast(v16bf, p);
}

// ---------------------------------------------------------------------------
// Kernel 0: zero dense-output region + per-expert counters
// ---------------------------------------------------------------------------
__global__ __launch_bounds__(256)
void moe_zero_kernel(float* __restrict__ out, int n, int* __restrict__ cnt) {
  int i = blockIdx.x * 256 + threadIdx.x;
  if (i < n) out[i] = 0.0f;
  if (blockIdx.x == 0 && threadIdx.x < E_) cnt[threadIdx.x] = 0;
}

// ---------------------------------------------------------------------------
// Kernel 1: router
// ---------------------------------------------------------------------------
__global__ __launch_bounds__(256)
void moe_router_kernel(const float* __restrict__ x, const float* __restrict__ Wr,
                       const float* __restrict__ br, float* __restrict__ probs_out,
                       int* __restrict__ cnt, int* __restrict__ list,
                       float* __restrict__ wlist) {
  __shared__ float sWr[D_ * E_];
  const int tid = threadIdx.x;
  for (int i = 0; i < (D_ * E_) / 256; ++i) sWr[tid + i * 256] = Wr[tid + i * 256];
  __syncthreads();

  const int t = blockIdx.x * 256 + tid;
  const float* xr = x + (size_t)t * D_;

  float acc[E_];
  #pragma unroll
  for (int e = 0; e < E_; ++e) acc[e] = br[e];

  for (int d = 0; d < D_; d += 4) {
    float4 xv = *(const float4*)(xr + d);
    #pragma unroll
    for (int e = 0; e < E_; ++e) {
      acc[e] += xv.x * sWr[(d + 0) * E_ + e] + xv.y * sWr[(d + 1) * E_ + e]
              + xv.z * sWr[(d + 2) * E_ + e] + xv.w * sWr[(d + 3) * E_ + e];
    }
  }

  float m = acc[0];
  #pragma unroll
  for (int e = 1; e < E_; ++e) m = fmaxf(m, acc[e]);
  float p[E_], sum = 0.0f;
  #pragma unroll
  for (int e = 0; e < E_; ++e) { p[e] = __expf(acc[e] - m); sum += p[e]; }
  float inv = 1.0f / sum;
  #pragma unroll
  for (int e = 0; e < E_; ++e) { p[e] *= inv; probs_out[(size_t)t * E_ + e] = p[e]; }

  int i1 = 0;
  #pragma unroll
  for (int e = 1; e < E_; ++e) if (p[e] > p[i1]) i1 = e;
  int i2 = (i1 == 0) ? 1 : 0;
  #pragma unroll
  for (int e = 0; e < E_; ++e) if (e != i1 && p[e] > p[i2]) i2 = e;

  float s2 = p[i1] + p[i2];
  float c1 = p[i1] / s2, c2 = p[i2] / s2;

  int pos1 = atomicAdd(&cnt[i1], 1);
  list[i1 * NTOK + pos1] = t;  wlist[i1 * NTOK + pos1] = c1;
  int pos2 = atomicAdd(&cnt[i2], 1);
  list[i2 * NTOK + pos2] = t;  wlist[i2 * NTOK + pos2] = c2;
}

// ---------------------------------------------------------------------------
// Kernel T: batched transpose + f32->bf16.  in[e][R][C] f32 -> out[e][C][R] bf16.
// 64x64 LDS tiles, coalesced reads and (packed) writes.
// ---------------------------------------------------------------------------
__global__ __launch_bounds__(256)
void transpose_cvt_kernel(const float* __restrict__ in, unsigned short* __restrict__ outT,
                          int R, int C) {
  __shared__ float tile[64][65];
  const int ct     = blockIdx.x;             // tile along C
  const int rtiles = R >> 6;
  const int e      = blockIdx.y / rtiles;
  const int rt     = blockIdx.y % rtiles;
  const float*    inE  = in   + (size_t)e * R * C;
  unsigned short* outE = outT + (size_t)e * R * C;
  const int tid = threadIdx.x;
  const int tr  = tid >> 4;          // 0..15
  const int tc  = (tid & 15) * 4;    // 0..60

  #pragma unroll
  for (int i = 0; i < 4; ++i) {
    int r = tr + i * 16;
    float4 v = *(const float4*)(inE + (size_t)(rt * 64 + r) * C + ct * 64 + tc);
    tile[r][tc + 0] = v.x; tile[r][tc + 1] = v.y;
    tile[r][tc + 2] = v.z; tile[r][tc + 3] = v.w;
  }
  __syncthreads();
  #pragma unroll
  for (int i = 0; i < 4; ++i) {
    int c = tr + i * 16;               // out row = original column
    uint2 pk;
    pk.x = pk_bf16(tile[tc + 0][c], tile[tc + 1][c]);
    pk.y = pk_bf16(tile[tc + 2][c], tile[tc + 3][c]);
    *(uint2*)(outE + (size_t)(ct * 64 + c) * R + rt * 64 + tc) = pk;
  }
}

// ---------------------------------------------------------------------------
// Kernel 2 (fast): expert MLP on pre-converted n-major bf16 weights.
// B-fragments load DIRECTLY from global (16 contiguous bf16 per lane = two
// aligned global_load_b128) -> zero staging, zero k-loop barriers, zero
// conversion VALU. Weights (128MB bf16) are L2-resident on MI455X (192MB L2).
// ---------------------------------------------------------------------------
__global__ __launch_bounds__(256, 1)
void moe_expert_fast_kernel(const float* __restrict__ x,
                            const unsigned short* __restrict__ W1T,  // [E][H][D] bf16
                            const float* __restrict__ b1,
                            const unsigned short* __restrict__ W2T,  // [E][O][H] bf16
                            const float* __restrict__ b2,
                            const int* __restrict__ cnt, const int* __restrict__ list,
                            const float* __restrict__ wlist, float* __restrict__ out) {
  const int e    = blockIdx.x >> 7;
  const int tile = blockIdx.x & (TILES_PER_E - 1);
  const int m0   = tile * TM;
  const int nTok = cnt[e];
  if (m0 >= nTok) return;
  const int rows = (nTok - m0 < TM) ? (nTok - m0) : TM;

  extern __shared__ char smem[];
  unsigned short* Xs = (unsigned short*)smem;            // [TM][XS_LD]
  unsigned short* Hs = Xs + TM * XS_LD;                  // [TM][HS_LD]
  int*   tok = (int*)(Hs + TM * HS_LD);
  float* cw  = (float*)(tok + TM);

  const int tid  = threadIdx.x;
  const int lane = tid & 31;
  const int wave = tid >> 5;

  if (tid < TM) {
    if (tid < rows) {
      tok[tid] = list[e * NTOK + m0 + tid];
      cw[tid]  = wlist[e * NTOK + m0 + tid];
    } else { tok[tid] = -1; cw[tid] = 0.0f; }
  }
  __syncthreads();

  for (int i = 0; i < 16; ++i) {                         // gather x -> Xs bf16
    int idx = tid + i * 256;
    int r   = idx >> 8;
    int c4  = (idx & 255) * 4;
    float4 v = make_float4(0.f, 0.f, 0.f, 0.f);
    if (r < rows) v = *(const float4*)(x + (size_t)tok[r] * D_ + c4);
    uint2 pk = make_uint2(pk_bf16(v.x, v.y), pk_bf16(v.z, v.w));
    *(uint2*)(Xs + r * XS_LD + c4) = pk;
  }
  __syncthreads();

  const int arow  = lane & 15;
  const int kb_a  = (lane & 16) ? 16 : 0;
  const int kb_b  = (lane & 16) ? 16 : 0;
  const int nloc  = lane & 15;
  const int mbase = (lane & 16) ? 8 : 0;

  // -------------------- Phase A: Hs = GELU(Xs @ W1[e] + b1) --------------------
  const unsigned short* W1Te = W1T + (size_t)e * H_ * D_;
  for (int n0 = 0; n0 < H_; n0 += TN) {
    const int col0 = n0 + wave * 16 + nloc;
    const int col1 = col0 + 128;
    const unsigned short* bp0 = W1Te + (size_t)col0 * D_ + kb_b;
    const unsigned short* bp1 = W1Te + (size_t)col1 * D_ + kb_b;
    v8f acc0 = {}, acc1 = {};
    #pragma unroll 4
    for (int k0 = 0; k0 < D_; k0 += TK) {
      const unsigned short* ap = Xs + arow * XS_LD + k0 + kb_a;
      v16bf a  = make_frag(*(const v8u16*)ap, *(const v8u16*)(ap + 8));
      v16bf b0 = make_frag(*(const v8u16*)(bp0 + k0), *(const v8u16*)(bp0 + k0 + 8));
      v16bf b1f= make_frag(*(const v8u16*)(bp1 + k0), *(const v8u16*)(bp1 + k0 + 8));
      acc0 = __builtin_amdgcn_wmma_f32_16x16x32_bf16(false, a, false, b0,
                                                     (short)0, acc0, false, false);
      acc1 = __builtin_amdgcn_wmma_f32_16x16x32_bf16(false, a, false, b1f,
                                                     (short)0, acc1, false, false);
    }
    const float bias0 = b1[e * H_ + col0];
    const float bias1 = b1[e * H_ + col1];
    #pragma unroll
    for (int r = 0; r < 8; ++r) {
      float v0 = acc0[r] + bias0;
      float v1 = acc1[r] + bias1;
      float g0 = 0.5f * v0 * (1.0f + erff(v0 * 0.70710678118654752f));
      float g1 = 0.5f * v1 * (1.0f + erff(v1 * 0.70710678118654752f));
      Hs[(mbase + r) * HS_LD + col0] = f2bf(g0);
      Hs[(mbase + r) * HS_LD + col1] = f2bf(g1);
    }
  }
  __syncthreads();

  // ---------- Phase B: out[tok] += cw * (Hs @ W2[e] + b2)  (atomic) ----------
  const unsigned short* W2Te = W2T + (size_t)e * O_ * H_;
  for (int n0 = 0; n0 < O_; n0 += TN) {
    const int col0 = n0 + wave * 16 + nloc;
    const int col1 = col0 + 128;
    const unsigned short* bp0 = W2Te + (size_t)col0 * H_ + kb_b;
    const unsigned short* bp1 = W2Te + (size_t)col1 * H_ + kb_b;
    v8f acc0 = {}, acc1 = {};
    #pragma unroll 4
    for (int k0 = 0; k0 < H_; k0 += TK) {
      const unsigned short* ap = Hs + arow * HS_LD + k0 + kb_a;
      v16bf a  = make_frag(*(const v8u16*)ap, *(const v8u16*)(ap + 8));
      v16bf b0 = make_frag(*(const v8u16*)(bp0 + k0), *(const v8u16*)(bp0 + k0 + 8));
      v16bf b1f= make_frag(*(const v8u16*)(bp1 + k0), *(const v8u16*)(bp1 + k0 + 8));
      acc0 = __builtin_amdgcn_wmma_f32_16x16x32_bf16(false, a, false, b0,
                                                     (short)0, acc0, false, false);
      acc1 = __builtin_amdgcn_wmma_f32_16x16x32_bf16(false, a, false, b1f,
                                                     (short)0, acc1, false, false);
    }
    const float bias0 = b2[e * O_ + col0];
    const float bias1 = b2[e * O_ + col1];
    #pragma unroll
    for (int r = 0; r < 8; ++r) {
      int m = mbase + r;
      if (m < rows) {
        float s = cw[m];
        size_t base = (size_t)tok[m] * O_;
        atomicAdd(out + base + col0, (acc0[r] + bias0) * s);
        atomicAdd(out + base + col1, (acc1[r] + bias1) * s);
      }
    }
  }
}

// ---------------------------------------------------------------------------
// Kernel 2 (fallback, ws too small): LDS-staging version w/ on-the-fly convert
// ---------------------------------------------------------------------------
__global__ __launch_bounds__(256, 1)
void moe_expert_kernel(const float* __restrict__ x,  const float* __restrict__ W1,
                       const float* __restrict__ b1, const float* __restrict__ W2,
                       const float* __restrict__ b2, const int* __restrict__ cnt,
                       const int* __restrict__ list, const float* __restrict__ wlist,
                       float* __restrict__ out) {
  const int e    = blockIdx.x >> 7;
  const int tile = blockIdx.x & (TILES_PER_E - 1);
  const int m0   = tile * TM;
  const int nTok = cnt[e];
  if (m0 >= nTok) return;
  const int rows = (nTok - m0 < TM) ? (nTok - m0) : TM;

  extern __shared__ char smem[];
  unsigned short* Xs = (unsigned short*)smem;
  unsigned short* Hs = Xs + TM * XS_LD;
  unsigned short* Bs = Hs + TM * HS_LD;
  int*   tok = (int*)(Bs + 2 * BS_BUF);
  float* cw  = (float*)(tok + TM);

  const int tid  = threadIdx.x;
  const int lane = tid & 31;
  const int wave = tid >> 5;

  if (tid < TM) {
    if (tid < rows) {
      tok[tid] = list[e * NTOK + m0 + tid];
      cw[tid]  = wlist[e * NTOK + m0 + tid];
    } else { tok[tid] = -1; cw[tid] = 0.0f; }
  }
  __syncthreads();

  for (int i = 0; i < 16; ++i) {
    int idx = tid + i * 256;
    int r   = idx >> 8;
    int c4  = (idx & 255) * 4;
    float4 v = make_float4(0.f, 0.f, 0.f, 0.f);
    if (r < rows) v = *(const float4*)(x + (size_t)tok[r] * D_ + c4);
    uint2 pk = make_uint2(pk_bf16(v.x, v.y), pk_bf16(v.z, v.w));
    *(uint2*)(Xs + r * XS_LD + c4) = pk;
  }

  const int arow  = lane & 15;
  const int kb_a  = (lane & 16) ? 16 : 0;
  const int kb_b  = (lane & 16) ? 16 : 0;
  const int nloc  = lane & 15;
  const int mbase = (lane & 16) ? 8 : 0;

  unsigned short* bsme = Bs + tid * BS_LD;
  const unsigned short* bq0 = Bs + (wave * 16 + nloc) * BS_LD + kb_b;
  const unsigned short* bq1 = bq0 + 128 * BS_LD;

  float wreg[TK];

  const float* W1e = W1 + (size_t)e * D_ * H_;
  for (int n0 = 0; n0 < H_; n0 += TN) {
    const float* colp = W1e + n0 + tid;
    #pragma unroll
    for (int kk = 0; kk < TK; ++kk) wreg[kk] = colp[(size_t)kk * H_];
    v8f acc0 = {}, acc1 = {};
    #pragma unroll 2
    for (int i = 0; i < D_ / TK; ++i) {
      const int boff = (i & 1) * BS_BUF;
      #pragma unroll
      for (int c = 0; c < 4; ++c) {
        union { v8u16 v; unsigned int u[4]; } q;
        #pragma unroll
        for (int p = 0; p < 4; ++p)
          q.u[p] = pk_bf16(wreg[c * 8 + 2 * p], wreg[c * 8 + 2 * p + 1]);
        *(v8u16*)(bsme + boff + c * 8) = q.v;
      }
      if (i + 1 < D_ / TK) {
        const float* np = colp + (size_t)(i + 1) * TK * H_;
        #pragma unroll
        for (int kk = 0; kk < TK; ++kk) wreg[kk] = np[(size_t)kk * H_];
      }
      __syncthreads();
      const int k0 = i * TK;
      const unsigned short* ap = Xs + arow * XS_LD + k0 + kb_a;
      v16bf a  = make_frag(*(const v8u16*)ap, *(const v8u16*)(ap + 8));
      v16bf b0 = make_frag(*(const v8u16*)(bq0 + boff), *(const v8u16*)(bq0 + boff + 8));
      v16bf b1f= make_frag(*(const v8u16*)(bq1 + boff), *(const v8u16*)(bq1 + boff + 8));
      acc0 = __builtin_amdgcn_wmma_f32_16x16x32_bf16(false, a, false, b0,
                                                     (short)0, acc0, false, false);
      acc1 = __builtin_amdgcn_wmma_f32_16x16x32_bf16(false, a, false, b1f,
                                                     (short)0, acc1, false, false);
    }
    const int col0 = n0 + wave * 16 + nloc;
    const int col1 = col0 + 128;
    const float bias0 = b1[e * H_ + col0];
    const float bias1 = b1[e * H_ + col1];
    #pragma unroll
    for (int r = 0; r < 8; ++r) {
      float v0 = acc0[r] + bias0;
      float v1 = acc1[r] + bias1;
      float g0 = 0.5f * v0 * (1.0f + erff(v0 * 0.70710678118654752f));
      float g1 = 0.5f * v1 * (1.0f + erff(v1 * 0.70710678118654752f));
      Hs[(mbase + r) * HS_LD + col0] = f2bf(g0);
      Hs[(mbase + r) * HS_LD + col1] = f2bf(g1);
    }
  }
  __syncthreads();

  const float* W2e = W2 + (size_t)e * H_ * O_;
  for (int n0 = 0; n0 < O_; n0 += TN) {
    const float* colp = W2e + n0 + tid;
    #pragma unroll
    for (int kk = 0; kk < TK; ++kk) wreg[kk] = colp[(size_t)kk * O_];
    v8f acc0 = {}, acc1 = {};
    #pragma unroll 2
    for (int i = 0; i < H_ / TK; ++i) {
      const int boff = (i & 1) * BS_BUF;
      #pragma unroll
      for (int c = 0; c < 4; ++c) {
        union { v8u16 v; unsigned int u[4]; } q;
        #pragma unroll
        for (int p = 0; p < 4; ++p)
          q.u[p] = pk_bf16(wreg[c * 8 + 2 * p], wreg[c * 8 + 2 * p + 1]);
        *(v8u16*)(bsme + boff + c * 8) = q.v;
      }
      if (i + 1 < H_ / TK) {
        const float* np = colp + (size_t)(i + 1) * TK * O_;
        #pragma unroll
        for (int kk = 0; kk < TK; ++kk) wreg[kk] = np[(size_t)kk * O_];
      }
      __syncthreads();
      const int k0 = i * TK;
      const unsigned short* ap = Hs + arow * HS_LD + k0 + kb_a;
      v16bf a  = make_frag(*(const v8u16*)ap, *(const v8u16*)(ap + 8));
      v16bf b0 = make_frag(*(const v8u16*)(bq0 + boff), *(const v8u16*)(bq0 + boff + 8));
      v16bf b1f= make_frag(*(const v8u16*)(bq1 + boff), *(const v8u16*)(bq1 + boff + 8));
      acc0 = __builtin_amdgcn_wmma_f32_16x16x32_bf16(false, a, false, b0,
                                                     (short)0, acc0, false, false);
      acc1 = __builtin_amdgcn_wmma_f32_16x16x32_bf16(false, a, false, b1f,
                                                     (short)0, acc1, false, false);
    }
    const int col0 = n0 + wave * 16 + nloc;
    const int col1 = col0 + 128;
    const float bias0 = b2[e * O_ + col0];
    const float bias1 = b2[e * O_ + col1];
    #pragma unroll
    for (int r = 0; r < 8; ++r) {
      int m = mbase + r;
      if (m < rows) {
        float s = cw[m];
        size_t base = (size_t)tok[m] * O_;
        atomicAdd(out + base + col0, (acc0[r] + bias0) * s);
        atomicAdd(out + base + col1, (acc1[r] + bias1) * s);
      }
    }
  }
}

// ---------------------------------------------------------------------------
extern "C" void kernel_launch(void* const* d_in, const int* in_sizes, int n_in,
                              void* d_out, int out_size, void* d_ws, size_t ws_size,
                              hipStream_t stream) {
  (void)in_sizes; (void)n_in; (void)out_size;
  const float* x  = (const float*)d_in[0];
  const float* Wr = (const float*)d_in[1];
  const float* br = (const float*)d_in[2];
  const float* W1 = (const float*)d_in[3];
  const float* b1 = (const float*)d_in[4];
  const float* W2 = (const float*)d_in[5];
  const float* b2 = (const float*)d_in[6];

  float* out   = (float*)d_out;
  float* probs = out + (size_t)NTOK * O_;

  char*  ws    = (char*)d_ws;
  int*   cnt   = (int*)ws;
  int*   list  = (int*)(ws + WS_LIST_OFF);
  float* wlist = (float*)(ws + WS_WLIST_OFF);

  const int nOut = NTOK * O_;
  moe_zero_kernel<<<(nOut + 255) / 256, 256, 0, stream>>>(out, nOut, cnt);
  moe_router_kernel<<<NTOK / 256, 256, 0, stream>>>(x, Wr, br, probs, cnt, list, wlist);

  if (ws_size >= WS_NEED) {
    unsigned short* W1T = (unsigned short*)(ws + WS_W1T_OFF);   // [E][H][D] bf16
    unsigned short* W2T = (unsigned short*)(ws + WS_W2T_OFF);   // [E][O][H] bf16
    transpose_cvt_kernel<<<dim3(H_ / 64, (D_ / 64) * E_), 256, 0, stream>>>(W1, W1T, D_, H_);
    transpose_cvt_kernel<<<dim3(O_ / 64, (H_ / 64) * E_), 256, 0, stream>>>(W2, W2T, H_, O_);
    hipFuncSetAttribute(reinterpret_cast<const void*>(moe_expert_fast_kernel),
                        hipFuncAttributeMaxDynamicSharedMemorySize, LDS_BYTES_FAST);
    moe_expert_fast_kernel<<<E_ * TILES_PER_E, 256, LDS_BYTES_FAST, stream>>>(
        x, W1T, b1, W2T, b2, cnt, list, wlist, out);
  } else {
    hipFuncSetAttribute(reinterpret_cast<const void*>(moe_expert_kernel),
                        hipFuncAttributeMaxDynamicSharedMemorySize, LDS_BYTES_STAGE);
    moe_expert_kernel<<<E_ * TILES_PER_E, 256, LDS_BYTES_STAGE, stream>>>(
        x, W1, b1, W2, b2, cnt, list, wlist, out);
  }
}
